// HessianCompatibleMultiHeadAttention_16192026706464
// MI455X (gfx1250) — matrix operand order repaired and checked by reading, not compiled, not run on hardware
//
#include <hip/hip_runtime.h>
#include <hip/hip_bf16.h>

// ---------------------------------------------------------------------------
// CDNA5 (gfx1250) bf16-WMMA multi-head attention
//   x[2,2048,1024] fp32, W[1024,1024] torch-style [out,in], y = x @ W^T
// Pipeline (all bf16 WMMA, fp32 accum):
//   1) fp32 -> bf16 convert (x, w_q, w_k, w_v, w_o)
//   2) Q = x Wq^T, K = x Wk^T  -> bf16 [4096,1024]      (32x64 tiles, dbl-buf)
//      V = x Wv^T              -> bf16 transposed [b,h,dk,s]
//   3) flash attention, 4 waves/block share (b,h); K/V tiles double-buffered
//      in LDS via async global->LDS DMA; row-sum via ones-WMMA; row-max via
//      dpp8/permlane16 butterflies (no LDS bpermute)
//   4) out = ctx Wo^T + b_o    -> fp32
// ---------------------------------------------------------------------------

typedef __attribute__((ext_vector_type(16))) __bf16 v16bf;
typedef __attribute__((ext_vector_type(8)))  __bf16 v8bf;
typedef __attribute__((ext_vector_type(4)))  __bf16 v4bf;
typedef __attribute__((ext_vector_type(8)))  float  v8f;
typedef __attribute__((ext_vector_type(4)))  int    v4i;

#if defined(__gfx1250__) && __has_builtin(__builtin_amdgcn_global_load_async_to_lds_b128) && __has_builtin(__builtin_amdgcn_s_wait_asynccnt)
#define USE_ASYNC_LDS 1
#else
#define USE_ASYNC_LDS 0
#endif

#if defined(__gfx1250__) && __has_builtin(__builtin_amdgcn_mov_dpp8) && __has_builtin(__builtin_amdgcn_permlane16)
#define USE_LANE_OPS 1
#else
#define USE_LANE_OPS 0
#endif

#if USE_ASYNC_LDS
typedef __attribute__((address_space(1))) v4i* gptr4;   // global int4*
typedef __attribute__((address_space(3))) v4i* lptr4;   // LDS int4*
#endif

__device__ __forceinline__ v16bf ld16(const __bf16* p) {
    union { v16bf v; v8bf h[2]; } u;
    u.h[0] = *(const v8bf*)p;
    u.h[1] = *(const v8bf*)(p + 8);
    return u.v;
}
__device__ __forceinline__ v16bf ld16_2(const __bf16* p0, const __bf16* p1) {
    union { v16bf v; v8bf h[2]; } u;
    u.h[0] = *(const v8bf*)p0;
    u.h[1] = *(const v8bf*)p1;
    return u.v;
}
__device__ __forceinline__ v8f wmma_bf16(v16bf a, v16bf b, v8f c) {
    // D = A(16x32) * B(32x16) + C, fp32 accum
    return __builtin_amdgcn_wmma_f32_16x16x32_bf16(
        /*neg_a=*/false, a, /*neg_b=*/false, b,
        /*c_mod=*/(short)0, c, /*reuse_a=*/false, /*reuse_b=*/false);
}

// 16-byte global -> LDS staging (async DMA path when the builtin exists)
__device__ __forceinline__ void stage16(const __bf16* g, __bf16* l) {
#if USE_ASYNC_LDS
    __builtin_amdgcn_global_load_async_to_lds_b128(
        (gptr4)g, (lptr4)l, /*offset=*/0, /*cpol=*/0);
#else
    *(v8bf*)l = *(const v8bf*)g;
#endif
}

// max across the 16 lanes of each wave32 half-row (butterfly)
__device__ __forceinline__ float rowmax16(float x) {
#if USE_LANE_OPS
    union f2i { float f; int i; } u, v;
    // dpp8 xor1: [1,0,3,2,5,4,7,6] ; xor2: [2,3,0,1,6,7,4,5] ; xor4: [4,5,6,7,0,1,2,3]
    u.f = x; v.i = __builtin_amdgcn_mov_dpp8(u.i, 14570689); x = fmaxf(x, v.f);
    u.f = x; v.i = __builtin_amdgcn_mov_dpp8(u.i, 11788826); x = fmaxf(x, v.f);
    u.f = x; v.i = __builtin_amdgcn_mov_dpp8(u.i, 6852524);  x = fmaxf(x, v.f);
    // permlane16 xor8 within each 16-lane row
    u.f = x;
    v.i = __builtin_amdgcn_permlane16(u.i, u.i, 0xFEDCBA98u, 0x76543210u, false, false);
    x = fmaxf(x, v.f);
#else
    x = fmaxf(x, __shfl_xor(x, 1, 16));
    x = fmaxf(x, __shfl_xor(x, 2, 16));
    x = fmaxf(x, __shfl_xor(x, 4, 16));
    x = fmaxf(x, __shfl_xor(x, 8, 16));
#endif
    return x;
}

// ---------------------------------------------------------------- convert ---
__global__ void cvt_f32_bf16_x4(const float* __restrict__ in,
                                __bf16* __restrict__ out, int n) {
    int i = (blockIdx.x * 256 + threadIdx.x) * 4;   // n is a multiple of 4
    if (i < n) {
        float4 f = *(const float4*)(in + i);
        v4bf o;
        o.x = (__bf16)f.x; o.y = (__bf16)f.y; o.z = (__bf16)f.z; o.w = (__bf16)f.w;
        *(v4bf*)(out + i) = o;
    }
}

// ----------------------------------------------------------------- GEMM ----
// Y[row,col] = sum_k A[row,k] * W[col,k];  A:[4096,1024] W:[1024,1024] bf16.
// One wave computes a 32x64 tile (8 WMMAs / 32-k step); all fragments for
// step k+1 are loaded before step k's WMMAs so loads overlap the matrix pipe.
// MODE 0: Y bf16 row-major [4096,1024]
// MODE 1: Y bf16 head-transposed V: idx = (b*1024 + col)*2048 + s
// MODE 2: Y fp32 row-major + bias[col]
template <int MODE>
__global__ __launch_bounds__(128, 1)
void gemm32x64(const __bf16* __restrict__ A,
               const __bf16* __restrict__ W,
               void* __restrict__ Yv,
               const float* __restrict__ bias) {
    const int lane = threadIdx.x & 31;
    const int warp = threadIdx.x >> 5;
    const int tile = blockIdx.x * 4 + warp;   // 2048 tiles of 32x64
    const int rowBase = (tile >> 4) * 32;     // 128 row tiles
    const int colBase = (tile & 15) * 64;     // 16 col tiles
    const int ln = lane & 15;
    const int hi = lane >> 4;

    const __bf16* arow0 = A + (size_t)(rowBase + ln) * 1024;
    const __bf16* arow1 = arow0 + (size_t)16 * 1024;

    auto ldA = [&](const __bf16* base, int k) {
        const int c0 = k + hi * 16;
        return ld16(base + c0);
    };
    auto ldB = [&](int c, int k) {
        return ld16(W + (size_t)(colBase + c * 16 + ln) * 1024 + k + hi * 16);
    };

    v8f acc0[4] = {}, acc1[4] = {};
    v16bf a0 = ldA(arow0, 0), a1 = ldA(arow1, 0);
    v16bf b[4];
#pragma unroll
    for (int c = 0; c < 4; ++c) b[c] = ldB(c, 0);

    for (int k = 0; k < 1024; k += 32) {
        const int kn = (k + 32) & 1023;       // last iter harmlessly reloads k=0
        v16bf na0 = ldA(arow0, kn);
        v16bf na1 = ldA(arow1, kn);
        v16bf nb[4];
#pragma unroll
        for (int c = 0; c < 4; ++c) nb[c] = ldB(c, kn);
#pragma unroll
        for (int c = 0; c < 4; ++c) {
            acc0[c] = wmma_bf16(a0, b[c], acc0[c]);
            acc1[c] = wmma_bf16(a1, b[c], acc1[c]);
        }
        a0 = na0; a1 = na1;
#pragma unroll
        for (int c = 0; c < 4; ++c) b[c] = nb[c];
    }

#pragma unroll
    for (int half = 0; half < 2; ++half) {
        v8f* acc = half ? acc1 : acc0;
#pragma unroll
        for (int c = 0; c < 4; ++c) {
#pragma unroll
            for (int v = 0; v < 8; ++v) {
                const int row = rowBase + half * 16 + v + hi * 8;  // C/D layout
                const int col = colBase + c * 16 + ln;
                if (MODE == 0) {
                    ((__bf16*)Yv)[(size_t)row * 1024 + col] = (__bf16)acc[c][v];
                } else if (MODE == 1) {
                    const int bb = row >> 11, s = row & 2047;
                    ((__bf16*)Yv)[((size_t)(bb * 1024 + col) << 11) + s] =
                        (__bf16)acc[c][v];
                } else {
                    ((float*)Yv)[(size_t)row * 1024 + col] = acc[c][v] + bias[col];
                }
            }
        }
    }
}

// ------------------------------------------------------- flash attention ---
// Qb,Kb: bf16 [4096,1024] ([b,s,h*64+dk]);  Vt: bf16 [b,h,dk,2048];
// ctx: bf16 [4096,1024].
// 4 waves/block handle 4 query tiles of the SAME (b,h). K/V tiles for each
// 32-key block are staged cooperatively into double-buffered LDS: block i+1's
// async DMA is issued before computing block i, and s_wait_asynccnt(4) only
// drains the previous buffer (async loads complete in order). Rows padded
// (72 / 40 halves) so 16-lane fragment reads hit 16 distinct bank groups.
__global__ __launch_bounds__(128, 1)
void attn_flash(const __bf16* __restrict__ Qb,
                const __bf16* __restrict__ Kb,
                const __bf16* __restrict__ Vt,
                __bf16* __restrict__ ctx) {
    __shared__ __bf16 Ks[2][32][72];     // 2 x 4.5 KB, key-major [key][dk]
    __shared__ __bf16 Vs[2][64][40];     // 2 x 5.0 KB, dk-major  [dk][s]
    __shared__ __bf16 Ps[4][16][32];     // 4.0 KB, per-wave P tile

    const int tid  = threadIdx.x;
    const int lane = tid & 31;
    const int warp = tid >> 5;
    const int gid  = blockIdx.x;          // 1024 blocks
    const int b  = gid >> 9;
    const int h  = (gid >> 5) & 15;
    const int qt = (gid & 31) * 4 + warp; // 128 query tiles per (b,h)
    const int ln = lane & 15;
    const int hi = lane >> 4;
    const int qRow = b * 2048 + qt * 16;

    const __bf16* kbase = Kb + (size_t)(b * 2048) * 1024 + h * 64;
    const __bf16* vbase = Vt + ((size_t)(b * 16 + h) * 64) * 2048;

    // each thread moves 4 x 16B per block: 2 chunks of K, 2 chunks of V
    auto stageBlock = [&](int buf, int kb) {
#pragma unroll
        for (int j = 0; j < 2; ++j) {
            const int chunk = tid * 2 + j;          // 0..255
            const int kr = chunk >> 3, kc = chunk & 7;
            stage16(kbase + (size_t)(kb + kr) * 1024 + kc * 8, &Ks[buf][kr][kc * 8]);
            const int vr = chunk >> 2, vc = chunk & 3;
            stage16(vbase + (size_t)vr * 2048 + kb + vc * 8, &Vs[buf][vr][vc * 8]);
        }
    };

    // Q A-frags over dk (K-dim): loaded once, reused for all key blocks
    const __bf16* qrow = Qb + (size_t)(qRow + ln) * 1024 + h * 64;
    v16bf qa0 = ld16(qrow + hi * 16);       // dk 0..31
    v16bf qa1 = ld16(qrow + 32 + hi * 16);  // dk 32..63

    v16bf ones;
#pragma unroll
    for (int i = 0; i < 16; ++i) ones[i] = (__bf16)1.0f;

    v8f o[4] = {};
    float rmax[8], rsum[8];
#pragma unroll
    for (int v = 0; v < 8; ++v) { rmax[v] = -1e30f; rsum[v] = 0.0f; }

    stageBlock(0, 0);                     // prologue DMA for block 0

    for (int i = 0; i < 64; ++i) {
        const int kb  = i * 32;
        const int cur = i & 1;
        // issue next block's DMA into the other buffer (wraps harmlessly at end)
        stageBlock(cur ^ 1, (kb + 32) & 2047);
#if USE_ASYNC_LDS
        __builtin_amdgcn_s_wait_asynccnt(4);   // drain previous buffer only
#endif
        __syncthreads();                       // buf[cur] visible to all waves
        __builtin_prefetch(kbase + (size_t)((kb + 64) & 2047) * 1024, 0, 1);

        // ---- scores for 32 keys as two 16x16 tiles (K-dim = dk = 64) ------
        v8f s0 = {}, s1 = {};
        s0 = wmma_bf16(qa0, ld16(&Ks[cur][ln][hi * 16]), s0);
        s0 = wmma_bf16(qa1, ld16(&Ks[cur][ln][32 + hi * 16]), s0);
        s1 = wmma_bf16(qa0, ld16(&Ks[cur][16 + ln][hi * 16]), s1);
        s1 = wmma_bf16(qa1, ld16(&Ks[cur][16 + ln][32 + hi * 16]), s1);

        // ---- online softmax: max via lane ops, sum via ones-WMMA ----------
        float corr[8];
#pragma unroll
        for (int v = 0; v < 8; ++v) {
            float a0 = s0[v] * 0.125f;                // 1/sqrt(Dk)
            float a1 = s1[v] * 0.125f;
            float mx = rowmax16(fmaxf(a0, a1));
            float nm = fmaxf(rmax[v], mx);
            corr[v]  = __expf(rmax[v] - nm);
            rmax[v]  = nm;
            float p0 = __expf(a0 - nm);
            float p1 = __expf(a1 - nm);
#pragma unroll
            for (int c = 0; c < 4; ++c) o[c][v] *= corr[v];
            const int m = v + hi * 8;                 // C/D-layout row
            Ps[warp][m][ln]      = (__bf16)p0;        // P row-major 16x32
            Ps[warp][m][16 + ln] = (__bf16)p1;
        }
        // Re-read P as an A-frag (per-wave region; same-wave DS ops in order)
        v16bf pa = ld16(&Ps[warp][ln][hi * 16]);

        // row sums on the matrix pipe: P(16x32) x ones(32x16)
        v8f bs = {};
        bs = wmma_bf16(pa, ones, bs);
#pragma unroll
        for (int v = 0; v < 8; ++v) rsum[v] = rsum[v] * corr[v] + bs[v];

        // ---- O += P(16x32) * V(32x64) from the shared LDS V tile ----------
#pragma unroll
        for (int c = 0; c < 4; ++c)
            o[c] = wmma_bf16(pa, ld16(&Vs[cur][c * 16 + ln][hi * 16]), o[c]);

        __syncthreads();   // all waves done with buf[cur] before it is restaged
    }

    // normalize and store ctx (bf16, [b,s,h*64+dk] row-major)
#pragma unroll
    for (int v = 0; v < 8; ++v) {
        float inv = 1.0f / rsum[v];
        const int row = qRow + v + hi * 8;
#pragma unroll
        for (int c = 0; c < 4; ++c) {
            ctx[(size_t)row * 1024 + h * 64 + c * 16 + ln] = (__bf16)(o[c][v] * inv);
        }
    }
}

// ---------------------------------------------------------------- launch ---
extern "C" void kernel_launch(void* const* d_in, const int* in_sizes, int n_in,
                              void* d_out, int out_size, void* d_ws, size_t ws_size,
                              hipStream_t stream) {
    (void)in_sizes; (void)n_in; (void)out_size; (void)ws_size;
    const float* x   = (const float*)d_in[0];   // [2,2048,1024]
    const float* w_q = (const float*)d_in[1];   // [1024,1024]
    const float* w_k = (const float*)d_in[2];
    const float* w_v = (const float*)d_in[3];
    const float* w_o = (const float*)d_in[4];
    const float* b_o = (const float*)d_in[5];   // [1024]
    float* out = (float*)d_out;                 // [2,2048,1024]

    const int NX = 4096 * 1024;    // x / Q / K / V / ctx element counts
    const int NW = 1024 * 1024;    // weight element counts

    __bf16* wsb = (__bf16*)d_ws;   // 48 MB of bf16 scratch
    __bf16* xb  = wsb;
    __bf16* wqb = xb  + NX;
    __bf16* wkb = wqb + NW;
    __bf16* wvb = wkb + NW;
    __bf16* wob = wvb + NW;
    __bf16* Qb  = wob + NW;
    __bf16* Kb  = Qb  + NX;
    __bf16* Vt  = Kb  + NX;        // [b,h,dk,s]
    __bf16* ctx = Vt  + NX;

    // 1) fp32 -> bf16
    cvt_f32_bf16_x4<<<NX / 1024, 256, 0, stream>>>(x,   xb,  NX);
    cvt_f32_bf16_x4<<<NW / 1024, 256, 0, stream>>>(w_q, wqb, NW);
    cvt_f32_bf16_x4<<<NW / 1024, 256, 0, stream>>>(w_k, wkb, NW);
    cvt_f32_bf16_x4<<<NW / 1024, 256, 0, stream>>>(w_v, wvb, NW);
    cvt_f32_bf16_x4<<<NW / 1024, 256, 0, stream>>>(w_o, wob, NW);

    // 2) projections (V stored head-transposed); 2048 tiles / 4 waves
    gemm32x64<0><<<512, 128, 0, stream>>>(xb, wqb, Qb, nullptr);
    gemm32x64<0><<<512, 128, 0, stream>>>(xb, wkb, Kb, nullptr);
    gemm32x64<1><<<512, 128, 0, stream>>>(xb, wvb, Vt, nullptr);

    // 3) flash attention
    attn_flash<<<1024, 128, 0, stream>>>(Qb, Kb, Vt, ctx);

    // 4) output projection + bias (fp32 out)
    gemm32x64<2><<<512, 128, 0, stream>>>(ctx, wob, out, b_o);
}
